// ScaleDotProductAttention_23648089932002
// MI455X (gfx1250) — matrix operand order repaired and checked by reading, not compiled, not run on hardware
//
#include <hip/hip_runtime.h>

typedef __attribute__((ext_vector_type(16))) _Float16 v16h;
typedef __attribute__((ext_vector_type(8)))  _Float16 v8h;
typedef __attribute__((ext_vector_type(8)))  float    v8f;

constexpr int kB = 8;
constexpr int kT = 2048;
constexpr int kE = 512;
constexpr float kScale = 0.04419417382415922f; // 1/sqrt(512)

__device__ __forceinline__ v16h join16(v8h lo, v8h hi) {
    return __builtin_shufflevector(lo, hi, 0,1,2,3,4,5,6,7,8,9,10,11,12,13,14,15);
}
__device__ __forceinline__ v8h  ld8 (const _Float16* p) { return *(const v8h *)p; }
__device__ __forceinline__ v16h ld16(const _Float16* p) { return *(const v16h*)p; }

__device__ __forceinline__ v8f wmma_f16(v16h a, v16h b, v8f c) {
    return __builtin_amdgcn_wmma_f32_16x16x32_f16(false, a, false, b,
                                                  (short)0, c, false, false);
}

// ---------------------------------------------------------------------------
// Tiled transpose + f32->f16 convert: dst[cols][rows] = src[rows][cols]
// grid: (cols/64, rows/64, batch), 256 threads
// ---------------------------------------------------------------------------
__global__ __launch_bounds__(256, 1)
void transpose_cvt_kernel(const float* __restrict__ src, _Float16* __restrict__ dst,
                          int rows, int cols)
{
    __shared__ _Float16 tile[64][72];
    const size_t boff = (size_t)blockIdx.z * rows * cols;
    src += boff;  dst += boff;
    const int c0 = blockIdx.x * 64, r0 = blockIdx.y * 64;

    for (int i = threadIdx.x; i < 64 * 64; i += 256) {
        const int r = i >> 6, c = i & 63;
        tile[r][c] = (_Float16)src[(size_t)(r0 + r) * cols + c0 + c];
    }
    __syncthreads();
    for (int i = threadIdx.x; i < 64 * 64; i += 256) {
        const int c = i >> 6, r = i & 63;
        dst[(size_t)(c0 + c) * rows + r0 + r] = tile[r][c];
    }
}

// ---------------------------------------------------------------------------
// Kernel 1: q = (x @ Wq + bq) * scale ; k = x @ Wk + bk   (f16 outputs)
// ---------------------------------------------------------------------------
__global__ __launch_bounds__(256, 1)
void proj_qk_kernel(const float* __restrict__ x,
                    const _Float16* __restrict__ WqT, const float* __restrict__ bq,
                    const _Float16* __restrict__ WkT, const float* __restrict__ bk,
                    _Float16* __restrict__ qh, _Float16* __restrict__ kh)
{
    __shared__ _Float16 Xs[16][kE];

    const int rowBase = blockIdx.x * 16;
    const int tid  = threadIdx.x;
    const int wave = tid >> 5;
    const int lane = tid & 31;
    const int lmod = lane & 15;
    const int halfSel = lane >> 4;

    for (int i = tid; i < 16 * kE; i += 256) {
        const int r = i >> 9, c = i & (kE - 1);
        Xs[r][c] = (_Float16)x[(size_t)(rowBase + r) * kE + c];
    }
    __syncthreads();

    const bool isQ = (wave < 4);
    const _Float16* __restrict__ WT   = isQ ? WqT : WkT;
    const float*    __restrict__ bias = isQ ? bq : bk;
    _Float16*       __restrict__ outp = isQ ? qh : kh;
    const int colBase0 = (wave & 3) * 128;

    v8f acc[8] = {};
    for (int ks = 0; ks < kE; ks += 32) {
        const v16h a = join16(ld8(&Xs[lmod][ks +     16 * halfSel]),
                              ld8(&Xs[lmod][ks + 8 + 16 * halfSel]));
        #pragma unroll
        for (int ct = 0; ct < 8; ++ct) {
            const int col = colBase0 + ct * 16 + lmod;
            const v16h bm = ld16(&WT[(size_t)col * kE + ks + 16 * halfSel]);
            acc[ct] = wmma_f16(a, bm, acc[ct]);
        }
    }

    #pragma unroll
    for (int ct = 0; ct < 8; ++ct) {
        const int col = colBase0 + ct * 16 + lmod;
        const float bv = bias[col];
        #pragma unroll
        for (int r = 0; r < 8; ++r) {
            const int row = rowBase + r + 8 * halfSel;
            float v = acc[ct][r] + bv;
            if (isQ) v *= kScale;
            outp[(size_t)row * kE + col] = (_Float16)v;
        }
    }
}

// ---------------------------------------------------------------------------
// Kernel 2: flash attention, 64-key blocks. One workgroup = 16 query rows.
// 8 waves split E into 64-col slices. Q fragments hoisted out of the key
// loop; K/V fragments are direct 32B vector loads (L2-resident), with
// prefetch of the next block.
// ---------------------------------------------------------------------------
__global__ __launch_bounds__(256, 1)
void flash_attn_kernel(const _Float16* __restrict__ qh,
                       const _Float16* __restrict__ kh,
                       const _Float16* __restrict__ vT,   // [B][E][T] f16
                       float* __restrict__ out)
{
    __shared__ float    Spart[8][16][64];  // 32 KB per-wave partial scores
    __shared__ _Float16 Ps[16][64];        // 2 KB softmaxed P tile (f16)
    __shared__ float    mrow[16], lrow[16], arow[16];

    const int tilesPerBatch = kT / 16;
    const int bIdx  = blockIdx.x / tilesPerBatch;
    const int qBase = (blockIdx.x % tilesPerBatch) * 16;
    const int rowG0 = bIdx * kT + qBase;

    const int tid  = threadIdx.x;
    const int wave = tid >> 5;
    const int lane = tid & 31;
    const int lmod = lane & 15;
    const int halfSel = lane >> 4;
    const int eSlice = wave * 64;

    if (tid < 16) { mrow[tid] = -1.0e30f; lrow[tid] = 0.0f; }

    // hoisted Q A-fragments for this wave's E slice (loop invariant)
    const _Float16* __restrict__ qrow = qh + (size_t)(rowG0 + lmod) * kE;
    const v16h a0 = join16(ld8(qrow + eSlice +      16 * halfSel),
                           ld8(qrow + eSlice + 8 + 16 * halfSel));
    const v16h a1 = join16(ld8(qrow + eSlice + 32 + 16 * halfSel),
                           ld8(qrow + eSlice + 40 + 16 * halfSel));

    v8f acc[4] = {};

    for (int kRow = 0; kRow < kT; kRow += 64) {
        const _Float16* __restrict__ krow = kh + (size_t)(bIdx * kT + kRow + lmod) * kE;
        if (kRow + 64 < kT) {   // prefetch next key block (K and V rows)
            __builtin_prefetch(krow + (size_t)64 * kE, 0, 1);
            __builtin_prefetch(&vT[((size_t)bIdx * kE + eSlice + lmod) * kT + kRow + 64], 0, 1);
        }

        // ---- partial S = Q[:,slice] @ K[:,slice]^T  (16x64 per wave) ----
        v8f s[4] = {};
        #pragma unroll
        for (int nt = 0; nt < 4; ++nt) {         // 4 key tiles of 16
            const _Float16* kr = krow + (size_t)nt * 16 * kE;
            const v16h b0 = ld16(kr + eSlice +      16 * halfSel);
            const v16h b1 = ld16(kr + eSlice + 32 + 16 * halfSel);
            s[nt] = wmma_f16(a0, b0, s[nt]);
            s[nt] = wmma_f16(a1, b1, s[nt]);
        }
        #pragma unroll
        for (int nt = 0; nt < 4; ++nt)
            #pragma unroll
            for (int r = 0; r < 8; ++r)
                Spart[wave][r + 8 * halfSel][nt * 16 + lmod] = s[nt][r];
        __syncthreads();

        // ---- reduce partials + online softmax (one thread per q row) ----
        if (tid < 16) {
            float mx = mrow[tid];
            for (int j = 0; j < 64; ++j) {
                float v = 0.0f;
                #pragma unroll
                for (int w = 0; w < 8; ++w) v += Spart[w][tid][j];
                Spart[0][tid][j] = v;            // stash reduced row (own row: safe)
                mx = fmaxf(mx, v);
            }
            const float alpha = __expf(mrow[tid] - mx);
            float sum = 0.0f;
            for (int j = 0; j < 64; ++j) {
                const float p = __expf(Spart[0][tid][j] - mx);
                Ps[tid][j] = (_Float16)p;
                sum += p;
            }
            mrow[tid] = mx;
            lrow[tid] = lrow[tid] * alpha + sum;
            arow[tid] = alpha;
        }
        __syncthreads();

        // ---- O = O*alpha + P @ V ----
        const v16h ap0 = join16(ld8(&Ps[lmod][     16 * halfSel]),
                                ld8(&Ps[lmod][8 + 16 * halfSel]));
        const v16h ap1 = join16(ld8(&Ps[lmod][32 + 16 * halfSel]),
                                ld8(&Ps[lmod][40 + 16 * halfSel]));
        float af[8];
        #pragma unroll
        for (int r = 0; r < 8; ++r) af[r] = arow[r + 8 * halfSel];

        #pragma unroll
        for (int t = 0; t < 4; ++t) {
            const int cb = eSlice + t * 16 + lmod;
            const _Float16* vrow = &vT[((size_t)bIdx * kE + cb) * kT + kRow];
            const v16h bv0 = ld16(vrow +      16 * halfSel);
            const v16h bv1 = ld16(vrow + 32 + 16 * halfSel);
            #pragma unroll
            for (int r = 0; r < 8; ++r) acc[t][r] *= af[r];
            acc[t] = wmma_f16(ap0, bv0, acc[t]);
            acc[t] = wmma_f16(ap1, bv1, acc[t]);
        }
        // next iteration's Spart/Ps writes are ordered by the two barriers above
    }

    // ---- normalize and store ----
    #pragma unroll
    for (int t = 0; t < 4; ++t) {
        const int col = eSlice + t * 16 + lmod;
        #pragma unroll
        for (int r = 0; r < 8; ++r) {
            const int rl = r + 8 * halfSel;
            out[(size_t)(rowG0 + rl) * kE + col] = acc[t][r] / lrow[rl];
        }
    }
}

// ---------------------------------------------------------------------------
extern "C" void kernel_launch(void* const* d_in, const int* in_sizes, int n_in,
                              void* d_out, int out_size, void* d_ws, size_t ws_size,
                              hipStream_t stream)
{
    const float* x  = (const float*)d_in[0];
    const float* Wq = (const float*)d_in[1];
    const float* bq = (const float*)d_in[2];
    const float* Wk = (const float*)d_in[3];
    const float* bk = (const float*)d_in[4];
    float* out = (float*)d_out;

    // workspace layout (f16): qh 16MB | kh 16MB | vT 16MB | WqT 0.5MB | WkT 0.5MB
    char* ws = (char*)d_ws;
    const size_t xteBytes = (size_t)kB * kT * kE * sizeof(_Float16);
    const size_t wBytes   = (size_t)kE * kE * sizeof(_Float16);
    _Float16* qh  = (_Float16*)(ws);
    _Float16* kh  = (_Float16*)(ws + xteBytes);
    _Float16* vT  = (_Float16*)(ws + 2 * xteBytes);
    _Float16* WqT = (_Float16*)(ws + 3 * xteBytes);
    _Float16* WkT = (_Float16*)(ws + 3 * xteBytes + wBytes);

    transpose_cvt_kernel<<<dim3(kE / 64, kE / 64, 1), 256, 0, stream>>>(Wq, WqT, kE, kE);
    transpose_cvt_kernel<<<dim3(kE / 64, kE / 64, 1), 256, 0, stream>>>(Wk, WkT, kE, kE);
    transpose_cvt_kernel<<<dim3(kE / 64, kT / 64, kB), 256, 0, stream>>>(x, vT, kT, kE);

    const int nTiles = kB * kT / 16;   // 1024 workgroups
    proj_qk_kernel<<<nTiles, 256, 0, stream>>>(x, WqT, bq, WkT, bk, qh, kh);
    flash_attn_kernel<<<nTiles, 256, 0, stream>>>(qh, kh, vT, out);

    (void)in_sizes; (void)n_in; (void)out_size; (void)ws_size;
}
